// GNNAutoencoder_45904610459855
// MI455X (gfx1250) — compile-verified, matrix operand order checked by reading, not run on hardware
//
#include <hip/hip_runtime.h>
#include <hip/hip_bf16.h>

typedef __attribute__((ext_vector_type(16))) __bf16 v16bf;
typedef __attribute__((ext_vector_type(8)))  __bf16 v8bf;
typedef __attribute__((ext_vector_type(8)))  float  v8f;

#define BM 128
#define BN 128
#define BK 32
#define KPAD 40   // bf16 elems per LDS row (80B stride: 16B-aligned, conflict-light)

union V16U { v16bf v; v8bf h[2]; };

// C[M,N] = A[M,K] @ B[K,N] (+ bias[N]); A,B f32 in memory, converted to bf16
// in registers on the way to LDS; f32 accumulation via v_wmma_f32_16x16x32_bf16.
// Requires: M % 128 == 0, K % 32 == 0. N arbitrary (multiple of 4).
__global__ __launch_bounds__(256)
void gemm_bf16_wmma(const float* __restrict__ A, const float* __restrict__ B,
                    const float* __restrict__ bias, float* __restrict__ C,
                    int M, int N, int K)
{
  __shared__ __align__(16) __bf16 As[2][BM][KPAD];
  __shared__ __align__(16) __bf16 Bs[2][BN][KPAD];   // stored transposed: Bs[n][k]

  const int tid  = threadIdx.x;
  const int wave = tid >> 5;          // 0..7
  const int lane = tid & 31;
  const int half = lane >> 4;         // 0/1 (K-half selector per WMMA layout)
  const int l16  = lane & 15;
  const int waveM = wave & 3;         // 4 waves along M, 32 rows each
  const int waveN = wave >> 2;        // 2 waves along N, 64 cols each
  const int rowBase = blockIdx.y * BM;
  const int colBase = blockIdx.x * BN;
  const int nK = K / BK;

  // global-load work split (256 threads)
  const int ar = tid >> 3;            // A: row group 0..31
  const int ak = (tid & 7) << 2;      // A: k offset 0,4,...,28
  const int bn = (tid & 31) << 2;     // B: n offset 0,4,...,124
  const int bk = tid >> 5;            // B: k base 0..7

  v8f acc[2][4] = {};
  float4 ra[4], rb[4];

  auto loadG = [&](int kt) {
    const int k0 = kt * BK;
#pragma unroll
    for (int i = 0; i < 4; ++i) {
      int r = ar + 32 * i;                                  // 0..127
      ra[i] = *(const float4*)(A + (size_t)(rowBase + r) * K + k0 + ak);
    }
#pragma unroll
    for (int i = 0; i < 4; ++i) {
      int k = bk + 8 * i;                                   // 0..31
      int n = colBase + bn;
      if (n < N) rb[i] = *(const float4*)(B + (size_t)(k0 + k) * N + n);
      else       rb[i] = make_float4(0.f, 0.f, 0.f, 0.f);
    }
  };

  auto storeL = [&](int buf) {
#pragma unroll
    for (int i = 0; i < 4; ++i) {
      int r = ar + 32 * i;
      __bf16* p = &As[buf][r][ak];
      p[0] = (__bf16)ra[i].x; p[1] = (__bf16)ra[i].y;
      p[2] = (__bf16)ra[i].z; p[3] = (__bf16)ra[i].w;
    }
#pragma unroll
    for (int i = 0; i < 4; ++i) {                           // transpose into Bs[n][k]
      int k = bk + 8 * i;
      Bs[buf][bn + 0][k] = (__bf16)rb[i].x;
      Bs[buf][bn + 1][k] = (__bf16)rb[i].y;
      Bs[buf][bn + 2][k] = (__bf16)rb[i].z;
      Bs[buf][bn + 3][k] = (__bf16)rb[i].w;
    }
  };

  loadG(0);
  storeL(0);
  __syncthreads();

  int cur = 0;
  for (int kt = 0; kt < nK; ++kt) {
    if (kt + 1 < nK) loadG(kt + 1);    // overlap global fetch with WMMA

    // Fragment loads: 16-bit A 16x32 layout -> two aligned 16B LDS reads/lane.
    V16U a[2], b[4];
#pragma unroll
    for (int m = 0; m < 2; ++m) {
      int r = waveM * 32 + m * 16 + l16;
      a[m].h[0] = *(const v8bf*)&As[cur][r][half * 8];       // K 0..7 / 8..15
      a[m].h[1] = *(const v8bf*)&As[cur][r][16 + half * 8];  // K 16..23 / 24..31
    }
#pragma unroll
    for (int n = 0; n < 4; ++n) {
      int c = waveN * 64 + n * 16 + l16;
      b[n].h[0] = *(const v8bf*)&Bs[cur][c][half * 8];
      b[n].h[1] = *(const v8bf*)&Bs[cur][c][16 + half * 8];
    }

#pragma unroll
    for (int m = 0; m < 2; ++m)
#pragma unroll
      for (int n = 0; n < 4; ++n)
        acc[m][n] = __builtin_amdgcn_wmma_f32_16x16x32_bf16(
            false, a[m].v, false, b[n].v, (short)0, acc[m][n], false, false);

    if (kt + 1 < nK) storeL(cur ^ 1);
    __syncthreads();
    cur ^= 1;
  }

  // Epilogue: C/D 16x16 f32 layout: VGPR r, lane -> (M = r + 8*half, N = l16)
#pragma unroll
  for (int n = 0; n < 4; ++n) {
    int col = colBase + waveN * 64 + n * 16 + l16;
    if (col < N) {
      float bv = bias ? bias[col] : 0.f;
#pragma unroll
      for (int m = 0; m < 2; ++m) {
        int rg = rowBase + waveM * 32 + m * 16 + half * 8;
#pragma unroll
        for (int r = 0; r < 8; ++r)
          C[(size_t)(rg + r) * N + col] = acc[m][n][r] + bv;
      }
    }
  }
}

// ---------------- graph / elementwise kernels ----------------

__global__ void init_deg(float* deg, int n) {
  int i = blockIdx.x * blockDim.x + threadIdx.x;
  if (i < n) deg[i] = 1.0f;                    // self-loop contribution
}

__global__ void count_deg(const int* __restrict__ ei, float* deg, int E) {
  int e = blockIdx.x * blockDim.x + threadIdx.x;
  if (e < E) atomicAdd(&deg[ei[E + e]], 1.0f); // col = targets
}

__global__ void finish_deg(float* deg, int n) {
  int i = blockIdx.x * blockDim.x + threadIdx.x;
  if (i < n) deg[i] = rsqrtf(deg[i]);          // deg >= 1 always
}

// outg[i][c] = H[i][c] * dinv[i]^2 + b_gcn[c]   (self-loop message + bias init)
__global__ void selfloop_bias(const float* __restrict__ H, const float* __restrict__ dinv,
                              const float* __restrict__ b, float* __restrict__ outg) {
  int i = blockIdx.x, c = threadIdx.x;
  float di = dinv[i];
  outg[(size_t)i * 256 + c] = H[(size_t)i * 256 + c] * di * di + b[c];
}

// outg[col] += H[row] * (dinv[row]*dinv[col]) for every edge (one block/edge)
__global__ void scatter_edges(const float* __restrict__ H, const int* __restrict__ ei,
                              const float* __restrict__ dinv, float* __restrict__ outg, int E) {
  int e = blockIdx.x, c = threadIdx.x;
  int r = ei[e], co = ei[E + e];
  float nrm = dinv[r] * dinv[co];
  atomicAdd(&outg[(size_t)co * 256 + c], H[(size_t)r * 256 + c] * nrm);
}

// latent = relu(outg) @ W_enc + b_enc ; hdec = relu(latent @ W_dec1 + b_dec1)
__global__ __launch_bounds__(256)
void mlp_mid(const float* __restrict__ outg,
             const float* __restrict__ We, const float* __restrict__ be,
             const float* __restrict__ Wd1, const float* __restrict__ bd1,
             float* __restrict__ latent, float* __restrict__ hdec) {
  __shared__ float hs[256];
  __shared__ float ls[64];
  int i = blockIdx.x, t = threadIdx.x;
  hs[t] = fmaxf(outg[(size_t)i * 256 + t], 0.f);
  __syncthreads();
  if (t < 64) {
    float s = be[t];
#pragma unroll 8
    for (int k = 0; k < 256; ++k) s += hs[k] * We[k * 64 + t];
    ls[t] = s;
    latent[(size_t)i * 64 + t] = s;            // latent output is pre-ReLU
  }
  __syncthreads();
  float s = bd1[t];
#pragma unroll 8
  for (int k = 0; k < 64; ++k) s += ls[k] * Wd1[k * 256 + t];
  hdec[(size_t)i * 256 + t] = fmaxf(s, 0.f);
}

// ---------------- launcher ----------------

extern "C" void kernel_launch(void* const* d_in, const int* in_sizes, int n_in,
                              void* d_out, int out_size, void* d_ws, size_t ws_size,
                              hipStream_t stream) {
  const float* x   = (const float*)d_in[0];
  const int*   ei  = (const int*)  d_in[1];
  const float* Wg  = (const float*)d_in[2];
  const float* bg  = (const float*)d_in[3];
  const float* We  = (const float*)d_in[4];
  const float* be  = (const float*)d_in[5];
  const float* Wd1 = (const float*)d_in[6];
  const float* bd1 = (const float*)d_in[7];
  const float* Wd2 = (const float*)d_in[8];
  const float* bd2 = (const float*)d_in[9];

  const int N = 8192, E = 262144, DIN = 20000, DH = 256;

  // workspace layout
  float* dinv = (float*)d_ws;                               // N f32 (deg -> rsqrt in place)
  float* H    = (float*)((char*)d_ws + (64 << 10));         // N*DH f32
  float* outg = H + (size_t)N * DH;                         // N*DH f32
  float* hdec = outg + (size_t)N * DH;                      // N*DH f32

  float* recon  = (float*)d_out;                            // N*DIN f32
  float* latent = recon + (size_t)N * DIN;                  // N*64 f32

  // symmetric GCN normalization terms
  init_deg  <<<(N + 255) / 256, 256, 0, stream>>>(dinv, N);
  count_deg <<<(E + 255) / 256, 256, 0, stream>>>(ei, dinv, E);
  finish_deg<<<(N + 255) / 256, 256, 0, stream>>>(dinv, N);

  // H = x @ W_gcn   (8192 x 20000 x 256)
  dim3 g1((DH + BN - 1) / BN, N / BM);
  gemm_bf16_wmma<<<g1, 256, 0, stream>>>(x, Wg, nullptr, H, N, DH, DIN);

  // aggregate: self-loops + bias init, then edge scatter-add
  selfloop_bias<<<N, DH, 0, stream>>>(H, dinv, bg, outg);
  scatter_edges<<<E, DH, 0, stream>>>(H, ei, dinv, outg, E);

  // encoder tail + decoder head (tiny GEMMs, fused per node)
  mlp_mid<<<N, DH, 0, stream>>>(outg, We, be, Wd1, bd1, latent, hdec);

  // recon = hdec @ W_dec2 + b_dec2   (8192 x 256 x 20000)
  dim3 g2((DIN + BN - 1) / BN, N / BM);
  gemm_bf16_wmma<<<g2, 256, 0, stream>>>(hdec, Wd2, bd2, recon, N, DIN, DH);
}
